// MultiHeadedAttention_27324581937530
// MI455X (gfx1250) — matrix operand order repaired and checked by reading, not compiled, not run on hardware
//
#include <hip/hip_runtime.h>
#include <math.h>
#include <stdint.h>

// ---------------------------------------------------------------------------
// MI455X (gfx1250) multi-head attention forward.
// B=4, S=2048, D=1024, H=16, hd=64.  All matmuls on v_wmma_f32_16x16x32_f16.
// ~137 GFLOP vs ~120 MB HBM traffic -> compute bound; all f16 intermediates
// (64 MB workspace) are L2-resident (192 MB), attention streams from L2/LDS.
// Workspace:
//   q16  [B*H][S][64]  f16
//   k16  [B*H][S][64]  f16
//   vt16 [B*H][64][S]  f16   (V transposed so PV B-fragments are contiguous)
//   a16  [B*S][1024]   f16   (attention output, pre out-projection)
// ---------------------------------------------------------------------------

typedef __attribute__((ext_vector_type(8)))  _Float16 v8h;
typedef __attribute__((ext_vector_type(16))) _Float16 v16h;
typedef __attribute__((ext_vector_type(8)))  float    v8f;

#define SDIM   2048
#define DDIM   1024
#define HEADS  16
#define HD     64
#define MROWS  8192            // B*S
#define SG     72              // GEMM LDS stride (halfs): 144B = 9*16B
#define SP     40              // P-bounce LDS stride (halfs): 80B = 5*16B

// Build a 16-half WMMA fragment from two 16B-aligned 8-half runs.
__device__ __forceinline__ v16h ld16(const _Float16* p0, const _Float16* p1) {
  v8h lo = *(const v8h*)p0;
  v8h hi = *(const v8h*)p1;
  v16h r;
#pragma unroll
  for (int i = 0; i < 8; ++i) { r[i] = lo[i]; r[i + 8] = hi[i]; }
  return r;
}

// Pack two f32 -> two f16 in one u32 (enables b32 LDS stores).
__device__ __forceinline__ uint32_t pack2h(float x, float y) {
  union { _Float16 h[2]; uint32_t u; } t;
  t.h[0] = (_Float16)x; t.h[1] = (_Float16)y;
  return t.u;
}

// LDS byte offset of a __shared__ object (AS3->flat cast keeps offset in low 32).
__device__ __forceinline__ uint32_t lds_off(const void* p) {
  return (uint32_t)(uintptr_t)p;
}

// One 512B async chunk: 32 lanes x 16B, memory -> LDS, tracked by ASYNCcnt.
__device__ __forceinline__ void async_b128(uint32_t ldsoff, const void* gaddr) {
  asm volatile("global_load_async_to_lds_b128 %0, %1, off"
               :: "v"(ldsoff), "v"((unsigned long long)(uintptr_t)gaddr)
               : "memory");
}

// ---------------------------------------------------------------------------
// Kernel 1: Y = X(f32)[8192x1024] * W(f32)[1024x1024] + bias, emitted as f16.
// mode 0: dst[((b*H+h)*S + s)*64 + d]   (Q / K layout)
// mode 1: dst[((b*H+h)*64 + d)*S + s]   (V transposed layout)
// Block tile 128x64, 8 waves 4x2, wave tile 32x32, K-step 64 (8 WMMA/stage).
// ---------------------------------------------------------------------------
__global__ __launch_bounds__(256)
void qkv_gemm_f16(const float* __restrict__ X, const float* __restrict__ W,
                  const float* __restrict__ bias, _Float16* __restrict__ dst,
                  int mode) {
  __shared__ _Float16 ldsA[128 * SG];  // 128 rows x 64 k
  __shared__ _Float16 ldsB[64 * SG];   // 64 n-rows x 64 k (W transposed)

  const int tid  = threadIdx.x;
  const int lane = tid & 31;
  const int wid  = tid >> 5;
  const int hi4  = lane >> 4;
  const int lm   = lane & 15;
  const int m0   = blockIdx.y * 128;
  const int n0   = blockIdx.x * 64;
  const int wm   = wid >> 1;
  const int wn   = wid & 1;

  v8f acc[2][2] = {};

  for (int k0 = 0; k0 < DDIM; k0 += 64) {
    __syncthreads();
    // Stage A tile (128x64 f32 -> f16), packed u32 stores.
    {
      const int r  = tid >> 1;
      const int c0 = (tid & 1) * 32;
      const float4* src = (const float4*)(X + (size_t)(m0 + r) * DDIM + k0 + c0);
      uint32_t* d = (uint32_t*)(ldsA + r * SG + c0);
#pragma unroll
      for (int v = 0; v < 8; ++v) {
        float4 f = src[v];
        d[v * 2 + 0] = pack2h(f.x, f.y);
        d[v * 2 + 1] = pack2h(f.z, f.w);
      }
    }
    // Stage W tile transposed: k-pairs packed -> single b32 store per (n, kp).
    {
      const int kp = (tid >> 3) * 2;    // 0,2,..,62
      const int n8 = (tid & 7) * 8;     // 0..56
      const float4* s0 = (const float4*)(W + (size_t)(k0 + kp)     * DDIM + n0 + n8);
      const float4* s1 = (const float4*)(W + (size_t)(k0 + kp + 1) * DDIM + n0 + n8);
#pragma unroll
      for (int v = 0; v < 2; ++v) {
        float4 a0 = s0[v], a1 = s1[v];
        *(uint32_t*)(ldsB + (n8 + v * 4 + 0) * SG + kp) = pack2h(a0.x, a1.x);
        *(uint32_t*)(ldsB + (n8 + v * 4 + 1) * SG + kp) = pack2h(a0.y, a1.y);
        *(uint32_t*)(ldsB + (n8 + v * 4 + 2) * SG + kp) = pack2h(a0.z, a1.z);
        *(uint32_t*)(ldsB + (n8 + v * 4 + 3) * SG + kp) = pack2h(a0.w, a1.w);
      }
    }
    __syncthreads();

    v16h a[2][2], b[2][2];
#pragma unroll
    for (int mi = 0; mi < 2; ++mi)
#pragma unroll
      for (int kj = 0; kj < 2; ++kj) {
        const _Float16* p =
            ldsA + (wm * 32 + mi * 16 + lm) * SG + kj * 32 + hi4 * 8;
        a[mi][kj] = ld16(p, p + 16);
      }
#pragma unroll
    for (int ni = 0; ni < 2; ++ni)
#pragma unroll
      for (int kj = 0; kj < 2; ++kj) {
        const _Float16* p =
            ldsB + (wn * 32 + ni * 16 + lm) * SG + kj * 32 + hi4 * 8;
        b[ni][kj] = ld16(p, p + 16);
      }
#pragma unroll
    for (int mi = 0; mi < 2; ++mi)
#pragma unroll
      for (int ni = 0; ni < 2; ++ni) {
        acc[mi][ni] = __builtin_amdgcn_wmma_f32_16x16x32_f16(
            false, a[mi][0], false, b[ni][0], (short)0, acc[mi][ni], false, false);
        acc[mi][ni] = __builtin_amdgcn_wmma_f32_16x16x32_f16(
            false, a[mi][1], false, b[ni][1], (short)0, acc[mi][ni], false, false);
      }
  }

  // Epilogue: +bias, convert, store in head-reshaped f16 layout.
#pragma unroll
  for (int mi = 0; mi < 2; ++mi)
#pragma unroll
    for (int ni = 0; ni < 2; ++ni) {
      const int col = n0 + wn * 32 + ni * 16 + lm;
      const float bval = bias[col];
      const int h_ = col >> 6, d_ = col & 63;
#pragma unroll
      for (int r = 0; r < 8; ++r) {
        const int row = m0 + wm * 32 + mi * 16 + hi4 * 8 + r;
        const int b_ = row >> 11, s_ = row & 2047;
        const float val = acc[mi][ni][r] + bval;
        size_t off;
        if (mode == 0) off = ((size_t)(b_ * HEADS + h_) * SDIM + s_) * HD + d_;
        else           off = ((size_t)(b_ * HEADS + h_) * HD + d_) * SDIM + s_;
        dst[off] = (_Float16)val;
      }
    }
}

// ---------------------------------------------------------------------------
// Kernel 2: flash attention, workgroup-cooperative.
// 8 waves/block share one (b,h); K/Vt tiles (4KB each per 32-key step) are
// double-buffered in LDS via global_load_async_to_lds_b128 (one 512B chunk
// per wave per tensor), overlapped with WMMA compute, synced by ASYNCcnt.
// ---------------------------------------------------------------------------
__global__ __launch_bounds__(256)
void flash_attn(const _Float16* __restrict__ q16, const _Float16* __restrict__ k16,
                const _Float16* __restrict__ vt16, _Float16* __restrict__ a16) {
  __shared__ _Float16 ldsK[2][32 * HD];   // 32 keys x 64 d   (per buffer 4KB)
  __shared__ _Float16 ldsV[2][HD * 32];   // 64 d x 32 tokens (per buffer 4KB)
  __shared__ _Float16 ldsP[8][16 * SP];   // per-wave P transpose bounce

  const int tid  = threadIdx.x;
  const int lane = tid & 31;
  const int wid  = tid >> 5;
  const int hi4  = lane >> 4;
  const int lm   = lane & 15;

  const int bh = blockIdx.x >> 4;               // 0..63 : (b*H + h)
  const int q0 = ((blockIdx.x & 15) * 8 + wid) * 16;

  const size_t qk_base = (size_t)bh * SDIM * HD;
  const size_t vt_base = (size_t)bh * HD * SDIM;

  const uint32_t kOff[2] = { lds_off(&ldsK[0][0]), lds_off(&ldsK[1][0]) };
  const uint32_t vOff[2] = { lds_off(&ldsV[0][0]), lds_off(&ldsV[1][0]) };

  // Loop-invariant Q fragments (d = 0..31 and 32..63).
  v16h aQ[2];
#pragma unroll
  for (int j = 0; j < 2; ++j) {
    const _Float16* p = q16 + qk_base + (size_t)(q0 + lm) * HD + j * 32 + hi4 * 8;
    aQ[j] = ld16(p, p + 16);
  }

  float mrow[8], lrow[8];
#pragma unroll
  for (int r = 0; r < 8; ++r) { mrow[r] = -1e30f; lrow[r] = 0.0f; }
  v8f O[4] = {};
  const float scale = 0.125f;  // 1/sqrt(64)
  _Float16* myP = &ldsP[wid][0];

  // Stage one K/Vt tile pair: wave copies its 512B slice of each.
  auto stage = [&](int buf, int kk) {
    // K tile: rows kk..kk+31 are 4KB contiguous in [S][64] layout.
    async_b128(kOff[buf] + wid * 512 + lane * 16,
               (const char*)(k16 + qk_base + (size_t)kk * HD) + wid * 512 + lane * 16);
    // Vt tile: 64 rows of 64B (token run), row stride S halfs.
    const int d = wid * 8 + (lane >> 2);
    async_b128(vOff[buf] + d * 64 + (lane & 3) * 16,
               (const char*)(vt16 + vt_base + (size_t)d * SDIM + kk) + (lane & 3) * 16);
  };

  stage(0, 0);

  for (int kk = 0; kk < SDIM; kk += 32) {
    const int cur = (kk >> 5) & 1;
    // My async chunks for buf[cur] have landed; barrier makes all 16 chunks
    // visible and guarantees every wave is done reading buf[cur^1].
    asm volatile("s_wait_asynccnt 0x0" ::: "memory");
    __syncthreads();
    if (kk + 32 < SDIM) stage(cur ^ 1, kk + 32);  // overlap with compute below

    // Scores: two 16x16 tiles (keys kk..+15, kk+16..+31), contraction over d.
    v8f st[2] = {};
#pragma unroll
    for (int t = 0; t < 2; ++t)
#pragma unroll
      for (int j = 0; j < 2; ++j) {
        const _Float16* p = &ldsK[cur][(t * 16 + lm) * HD + j * 32 + hi4 * 8];
        v16h bK = ld16(p, p + 16);
        st[t] = __builtin_amdgcn_wmma_f32_16x16x32_f16(
            false, aQ[j], false, bK, (short)0, st[t], false, false);
      }

    // Online softmax. C-layout: lane = column, VGPR r = row hi4*8+r, so row
    // reductions are butterflies within each 16-lane group (masks 1,2,4,8).
    float sc[8];
#pragma unroll
    for (int r = 0; r < 8; ++r) {
      float a0 = st[0][r] * scale, a1 = st[1][r] * scale;
      st[0][r] = a0; st[1][r] = a1;
      float tm = fmaxf(a0, a1);
      tm = fmaxf(tm, __shfl_xor(tm, 1));
      tm = fmaxf(tm, __shfl_xor(tm, 2));
      tm = fmaxf(tm, __shfl_xor(tm, 4));
      tm = fmaxf(tm, __shfl_xor(tm, 8));
      const float mnew = fmaxf(mrow[r], tm);
      sc[r] = __expf(mrow[r] - mnew);
      mrow[r] = mnew;
    }
#pragma unroll
    for (int r = 0; r < 8; ++r) {
      float p0 = __expf(st[0][r] - mrow[r]);
      float p1 = __expf(st[1][r] - mrow[r]);
      st[0][r] = p0; st[1][r] = p1;
      float s = p0 + p1;
      s += __shfl_xor(s, 1);
      s += __shfl_xor(s, 2);
      s += __shfl_xor(s, 4);
      s += __shfl_xor(s, 8);
      lrow[r] = lrow[r] * sc[r] + s;
#pragma unroll
      for (int ni = 0; ni < 4; ++ni) O[ni][r] *= sc[r];
    }

    // Transpose P: C-layout -> row-major 16x32 in LDS -> A-fragment.
#pragma unroll
    for (int r = 0; r < 8; ++r) {
      myP[(hi4 * 8 + r) * SP + lm]      = (_Float16)st[0][r];
      myP[(hi4 * 8 + r) * SP + 16 + lm] = (_Float16)st[1][r];
    }
    asm volatile("s_wait_dscnt 0x0" ::: "memory");  // same-wave LDS RAW fence
    const _Float16* pA = myP + lm * SP + hi4 * 8;
    v16h aP = ld16(pA, pA + 16);

    // O += P(16x32) * V(32x64) from the staged Vt tile.
#pragma unroll
    for (int ni = 0; ni < 4; ++ni) {
      const _Float16* p = &ldsV[cur][(ni * 16 + lm) * 32 + hi4 * 8];
      v16h bV = ld16(p, p + 16);
      O[ni] = __builtin_amdgcn_wmma_f32_16x16x32_f16(
          false, aP, false, bV, (short)0, O[ni], false, false);
    }
  }

  // Normalize and store merged-head layout [B*S][1024] f16.
  const int b_ = bh >> 4, h_ = bh & 15;
#pragma unroll
  for (int r = 0; r < 8; ++r) {
    const float inv = 1.0f / lrow[r];
    const size_t row = (size_t)b_ * SDIM + q0 + hi4 * 8 + r;
#pragma unroll
    for (int ni = 0; ni < 4; ++ni)
      a16[row * DDIM + h_ * 64 + ni * 16 + lm] = (_Float16)(O[ni][r] * inv);
  }
}

// ---------------------------------------------------------------------------
// Kernel 3: out = A16(f16)[8192x1024] * Wo(f32) + bo, f32 output.
// A fragments stream from L2; Wo staged transposed as f16 in LDS, K-step 64.
// ---------------------------------------------------------------------------
__global__ __launch_bounds__(256)
void out_proj(const _Float16* __restrict__ A16, const float* __restrict__ W,
              const float* __restrict__ bias, float* __restrict__ out) {
  __shared__ _Float16 ldsB[64 * SG];

  const int tid  = threadIdx.x;
  const int lane = tid & 31;
  const int wid  = tid >> 5;
  const int hi4  = lane >> 4;
  const int lm   = lane & 15;
  const int m0   = blockIdx.y * 128;
  const int n0   = blockIdx.x * 64;
  const int wm   = wid >> 1;
  const int wn   = wid & 1;

  v8f acc[2][2] = {};

  for (int k0 = 0; k0 < DDIM; k0 += 64) {
    __syncthreads();
    {
      const int kp = (tid >> 3) * 2;
      const int n8 = (tid & 7) * 8;
      const float4* s0 = (const float4*)(W + (size_t)(k0 + kp)     * DDIM + n0 + n8);
      const float4* s1 = (const float4*)(W + (size_t)(k0 + kp + 1) * DDIM + n0 + n8);
#pragma unroll
      for (int v = 0; v < 2; ++v) {
        float4 a0 = s0[v], a1 = s1[v];
        *(uint32_t*)(ldsB + (n8 + v * 4 + 0) * SG + kp) = pack2h(a0.x, a1.x);
        *(uint32_t*)(ldsB + (n8 + v * 4 + 1) * SG + kp) = pack2h(a0.y, a1.y);
        *(uint32_t*)(ldsB + (n8 + v * 4 + 2) * SG + kp) = pack2h(a0.z, a1.z);
        *(uint32_t*)(ldsB + (n8 + v * 4 + 3) * SG + kp) = pack2h(a0.w, a1.w);
      }
    }
    __syncthreads();

    v16h a[2][2], b[2][2];
#pragma unroll
    for (int mi = 0; mi < 2; ++mi)
#pragma unroll
      for (int kj = 0; kj < 2; ++kj) {
        const _Float16* p = A16 + (size_t)(m0 + wm * 32 + mi * 16 + lm) * DDIM +
                            k0 + kj * 32 + hi4 * 8;
        a[mi][kj] = ld16(p, p + 16);
      }
#pragma unroll
    for (int ni = 0; ni < 2; ++ni)
#pragma unroll
      for (int kj = 0; kj < 2; ++kj) {
        const _Float16* p =
            ldsB + (wn * 32 + ni * 16 + lm) * SG + kj * 32 + hi4 * 8;
        b[ni][kj] = ld16(p, p + 16);
      }
#pragma unroll
    for (int mi = 0; mi < 2; ++mi)
#pragma unroll
      for (int ni = 0; ni < 2; ++ni) {
        acc[mi][ni] = __builtin_amdgcn_wmma_f32_16x16x32_f16(
            false, a[mi][0], false, b[ni][0], (short)0, acc[mi][ni], false, false);
        acc[mi][ni] = __builtin_amdgcn_wmma_f32_16x16x32_f16(
            false, a[mi][1], false, b[ni][1], (short)0, acc[mi][ni], false, false);
      }
  }

#pragma unroll
  for (int mi = 0; mi < 2; ++mi)
#pragma unroll
    for (int ni = 0; ni < 2; ++ni) {
      const int col = n0 + wn * 32 + ni * 16 + lm;
      const float bval = bias[col];
#pragma unroll
      for (int r = 0; r < 8; ++r) {
        const size_t row = (size_t)(m0 + wm * 32 + mi * 16 + hi4 * 8 + r);
        out[row * DDIM + col] = acc[mi][ni][r] + bval;
      }
    }
}

// ---------------------------------------------------------------------------
extern "C" void kernel_launch(void* const* d_in, const int* in_sizes, int n_in,
                              void* d_out, int out_size, void* d_ws, size_t ws_size,
                              hipStream_t stream) {
  (void)in_sizes; (void)n_in; (void)out_size; (void)ws_size;
  const float* query = (const float*)d_in[0];
  const float* bank  = (const float*)d_in[1];
  const float* Wq    = (const float*)d_in[2];
  const float* bq    = (const float*)d_in[3];
  const float* Wk    = (const float*)d_in[4];
  const float* bk    = (const float*)d_in[5];
  const float* Wv    = (const float*)d_in[6];
  const float* bv    = (const float*)d_in[7];
  const float* Wo    = (const float*)d_in[8];
  const float* bo    = (const float*)d_in[9];
  float* out = (float*)d_out;

  const size_t tsz = (size_t)MROWS * DDIM;  // 8M f16 elements per tensor
  _Float16* q16  = (_Float16*)d_ws;
  _Float16* k16  = q16 + tsz;
  _Float16* vt16 = k16 + tsz;
  _Float16* a16  = vt16 + tsz;              // total 64 MB workspace

  dim3 blk(256);
  dim3 gridG(DDIM / 64, MROWS / 128);       // (16, 64)

  qkv_gemm_f16<<<gridG, blk, 0, stream>>>(query, Wq, bq, q16, 0);
  qkv_gemm_f16<<<gridG, blk, 0, stream>>>(bank,  Wk, bk, k16, 0);
  qkv_gemm_f16<<<gridG, blk, 0, stream>>>(bank,  Wv, bv, vt16, 1);

  // 64 (b,h) pairs x 16 blocks; 8 waves/block = 8 q-tiles of 16 rows each.
  flash_attn<<<dim3(1024), blk, 0, stream>>>(q16, k16, vt16, a16);

  out_proj<<<gridG, blk, 0, stream>>>(a16, Wo, bo, out);
}
